// BidirectionalAttention_35957466202237
// MI455X (gfx1250) — compile-verified, matrix operand order checked by reading, not run on hardware
//
#include <hip/hip_runtime.h>
#include <hip/hip_bf16.h>
#include <math.h>

typedef __attribute__((ext_vector_type(16))) __bf16 v16bf;
typedef __attribute__((ext_vector_type(8)))  float  v8f;

#define KB_  64     // batch
#define KC_  1024   // context length
#define KQ_  64     // question length
#define KE_  256    // embed dim
#define CTILE 128
#define NTHREADS 256

// K-index base for WMMA 16-bit A/B operand: element pair p (0..7), half-wave group g (0/1)
__device__ __forceinline__ int wmma_kbase(int p, int g) {
    return ((p < 4) ? (2 * p) : (2 * p + 8)) + 8 * g;
}

__device__ __forceinline__ v8f wmma_bf16(v16bf a, v16bf b, v8f c) {
    return __builtin_amdgcn_wmma_f32_16x16x32_bf16(false, a, false, b, (short)0, c, false, false);
}

// Kernel 1: per (batch, context-tile): sim GEMM -> softmax_q -> P@question GEMM,
// plus row-max(sim) written to scratch for the q2c path.
__global__ void bidaf_main_kernel(const float* __restrict__ context,
                                  const float* __restrict__ question,
                                  const float* __restrict__ wsv,
                                  float* __restrict__ c2q,
                                  float* __restrict__ rowm)
{
    extern __shared__ __align__(16) char smem[];
    __bf16* qbf  = (__bf16*)smem;                 // question row-major [q][e]   32KB
    __bf16* qtbf = qbf + KQ_ * KE_;               // question transposed [e][q]  32KB
    float*  simf = (float*)(qtbf + KE_ * KQ_);    // sim tile [CTILE][64] f32    32KB (P overlays it)
    float*  wq   = simf + CTILE * KQ_;            // 256
    float*  wc   = wq + KE_;                      // 256
    float*  wm   = wc + KE_;                      // 256
    float*  qproj = wm + KE_;                     // 64

    const int b   = blockIdx.y;
    const int c0  = blockIdx.x * CTILE;
    const int tid = threadIdx.x;
    const int lane = tid & 31;
    const int wave = tid >> 5;
    const int g  = lane >> 4;      // half-wave group
    const int ln = lane & 15;

    // ---- stage question (bf16, both layouts) and weight vectors ----
    const float* qsrc = question + (size_t)b * KQ_ * KE_;
    for (int i = tid; i < KQ_ * KE_; i += NTHREADS) {
        float v = qsrc[i];
        qbf[i] = (__bf16)v;
        int q = i >> 8, e = i & (KE_ - 1);
        qtbf[e * KQ_ + q] = (__bf16)v;
    }
    for (int i = tid; i < KE_; i += NTHREADS) {
        wq[i] = wsv[i];
        wc[i] = wsv[KE_ + i];
        wm[i] = wsv[2 * KE_ + i];
    }
    __syncthreads();

    // qproj[q] = question[q] . w_q
    if (tid < KQ_) {
        float s = 0.f;
        for (int e = 0; e < KE_; ++e) s += (float)qbf[tid * KE_ + e] * wq[e];
        qproj[tid] = s;
    }
    __syncthreads();

    // ---- GEMM1: sim_tile = (context_rows * w_m) @ question^T  (M=16/wave, N=64, K=256) ----
    const int myrow = c0 + wave * 16 + ln;
    const float* crow = context + ((size_t)b * KC_ + myrow) * KE_;

    v8f acc0 = {0,0,0,0,0,0,0,0};
    v8f acc1 = {0,0,0,0,0,0,0,0};
    v8f acc2 = {0,0,0,0,0,0,0,0};
    v8f acc3 = {0,0,0,0,0,0,0,0};

    for (int eb = 0; eb < KE_; eb += 32) {
        v16bf a;
        #pragma unroll
        for (int p = 0; p < 8; ++p) {
            int k = eb + wmma_kbase(p, g);
            a[2 * p]     = (__bf16)(crow[k]     * wm[k]);
            a[2 * p + 1] = (__bf16)(crow[k + 1] * wm[k + 1]);
        }
        v16bf b0, b1, b2, b3;
        #pragma unroll
        for (int p = 0; p < 8; ++p) {
            int k = eb + wmma_kbase(p, g);
            b0[2*p] = qbf[(ln     ) * KE_ + k]; b0[2*p+1] = qbf[(ln     ) * KE_ + k + 1];
            b1[2*p] = qbf[(16 + ln) * KE_ + k]; b1[2*p+1] = qbf[(16 + ln) * KE_ + k + 1];
            b2[2*p] = qbf[(32 + ln) * KE_ + k]; b2[2*p+1] = qbf[(32 + ln) * KE_ + k + 1];
            b3[2*p] = qbf[(48 + ln) * KE_ + k]; b3[2*p+1] = qbf[(48 + ln) * KE_ + k + 1];
        }
        acc0 = wmma_bf16(a, b0, acc0);
        acc1 = wmma_bf16(a, b1, acc1);
        acc2 = wmma_bf16(a, b2, acc2);
        acc3 = wmma_bf16(a, b3, acc3);
    }

    // spill sim tile to LDS (C/D layout: lane ln = col, vgpr r -> row r + 8g)
    #pragma unroll
    for (int r = 0; r < 8; ++r) {
        int row = wave * 16 + 8 * g + r;
        simf[row * KQ_ + ln]      = acc0[r];
        simf[row * KQ_ + 16 + ln] = acc1[r];
        simf[row * KQ_ + 32 + ln] = acc2[r];
        simf[row * KQ_ + 48 + ln] = acc3[r];
    }
    __syncthreads();

    // ---- softmax over q per row; cproj only shifts the stored row-max ----
    if (tid < CTILE) {
        const int row = tid;
        const float* cr = context + ((size_t)b * KC_ + c0 + row) * KE_;
        float cp = 0.f;
        for (int e = 0; e < KE_; ++e) cp += cr[e] * wc[e];

        float* srow = simf + row * KQ_;
        float m = -INFINITY;
        for (int q = 0; q < KQ_; ++q) m = fmaxf(m, srow[q] + qproj[q]);
        float s = 0.f;
        for (int q = 0; q < KQ_; ++q) s += __expf(srow[q] + qproj[q] - m);
        float inv = 1.0f / s;
        __bf16* prow = (__bf16*)srow;   // in-place overlay: halfword q lands in float q/2 (already read)
        for (int q = 0; q < KQ_; ++q) {
            float p = __expf(srow[q] + qproj[q] - m) * inv;
            prow[q] = (__bf16)p;
        }
        rowm[(size_t)b * KC_ + c0 + row] = m + cp;   // full sim row-max for q2c softmax
    }
    __syncthreads();

    // ---- GEMM2: out_tile = P(16x64) @ question(64x256) ----
    const __bf16* pb = (const __bf16*)simf;   // row stride = KQ_*2 halfwords = 128
    v16bf a0, a1;
    #pragma unroll
    for (int p = 0; p < 8; ++p) {
        int kb = wmma_kbase(p, g);
        int rb = (wave * 16 + ln) * (KQ_ * 2);
        a0[2*p] = pb[rb + kb];      a0[2*p+1] = pb[rb + kb + 1];
        a1[2*p] = pb[rb + 32 + kb]; a1[2*p+1] = pb[rb + 32 + kb + 1];
    }
    float* outb = c2q + ((size_t)b * KC_ + c0) * KE_;
    for (int et = 0; et < 16; ++et) {
        v16bf b0, b1;
        #pragma unroll
        for (int p = 0; p < 8; ++p) {
            int kb = wmma_kbase(p, g);
            int nb = (et * 16 + ln) * KQ_;
            b0[2*p] = qtbf[nb + kb];      b0[2*p+1] = qtbf[nb + kb + 1];
            b1[2*p] = qtbf[nb + 32 + kb]; b1[2*p+1] = qtbf[nb + 32 + kb + 1];
        }
        v8f acc = {0,0,0,0,0,0,0,0};
        acc = wmma_bf16(a0, b0, acc);
        acc = wmma_bf16(a1, b1, acc);
        #pragma unroll
        for (int r = 0; r < 8; ++r) {
            int row = wave * 16 + 8 * g + r;
            outb[(size_t)row * KE_ + et * 16 + ln] = acc[r];
        }
    }
}

// Kernel 2: q2c_att[b,e] = softmax_c(rowm[b,:]) . context[b,:,e]
__global__ void bidaf_q2c_kernel(const float* __restrict__ context,
                                 const float* __restrict__ rowm,
                                 float* __restrict__ q2c)
{
    __shared__ float wsh[KC_];
    __shared__ float red[NTHREADS];
    const int b = blockIdx.x;
    const int tid = threadIdx.x;

    float lm = -INFINITY;
    for (int c = tid; c < KC_; c += NTHREADS) {
        float v = rowm[(size_t)b * KC_ + c];
        wsh[c] = v;
        lm = fmaxf(lm, v);
    }
    red[tid] = lm; __syncthreads();
    for (int s = NTHREADS / 2; s > 0; s >>= 1) {
        if (tid < s) red[tid] = fmaxf(red[tid], red[tid + s]);
        __syncthreads();
    }
    float m = red[0]; __syncthreads();

    float ls = 0.f;
    for (int c = tid; c < KC_; c += NTHREADS) {
        float e = __expf(wsh[c] - m);
        wsh[c] = e;
        ls += e;
    }
    red[tid] = ls; __syncthreads();
    for (int s = NTHREADS / 2; s > 0; s >>= 1) {
        if (tid < s) red[tid] += red[tid + s];
        __syncthreads();
    }
    float inv = 1.0f / red[0]; __syncthreads();

    // thread tid handles embedding column e = tid (coalesced over context rows)
    float acc = 0.f;
    const float* cb = context + (size_t)b * KC_ * KE_;
    for (int c = 0; c < KC_; ++c) acc += wsh[c] * cb[(size_t)c * KE_ + tid];
    q2c[(size_t)b * KE_ + tid] = acc * inv;
}

extern "C" void kernel_launch(void* const* d_in, const int* in_sizes, int n_in,
                              void* d_out, int out_size, void* d_ws, size_t ws_size,
                              hipStream_t stream) {
    (void)in_sizes; (void)n_in; (void)out_size; (void)ws_size;
    const float* context  = (const float*)d_in[0];
    const float* question = (const float*)d_in[1];
    const float* wsv      = (const float*)d_in[2];
    float* out  = (float*)d_out;
    float* rowm = (float*)d_ws;                    // B*C floats = 256 KB scratch

    const size_t shbytes =
        (size_t)(KQ_ * KE_ + KE_ * KQ_) * sizeof(__bf16)   // qbf + qtbf
        + (size_t)(CTILE * KQ_) * sizeof(float)            // sim tile (P overlays)
        + (size_t)(3 * KE_ + KQ_) * sizeof(float);         // wq, wc, wm, qproj

    dim3 grid1(KC_ / CTILE, KB_);
    bidaf_main_kernel<<<grid1, NTHREADS, shbytes, stream>>>(context, question, wsv, out, rowm);

    float* q2c = out + (size_t)KB_ * KC_ * KE_;
    bidaf_q2c_kernel<<<KB_, NTHREADS, 0, stream>>>(context, rowm, q2c);
}
